// MultiHeadAttention_49409303773901
// MI455X (gfx1250) — matrix operand order repaired and checked by reading, not compiled, not run on hardware
//
#include <hip/hip_runtime.h>
#include <hip/hip_bf16.h>

// ---------------------------------------------------------------------------
// MI455X (gfx1250) multi-head attention forward.
// bf16 WMMA (v_wmma_f32_16x16x32_bf16), fp32 accumulate.
// Async global->LDS tile staging (GLOBAL_LOAD_ASYNC_TO_LDS_B128, ASYNCcnt)
// with double buffering so HBM/L2 fetch overlaps WMMA + softmax.
// Pipeline: f32->bf16 convert (+weight transpose) | Q/K/V projection GEMMs |
//           flash attention | output projection GEMM (fp32 out).
// ---------------------------------------------------------------------------

typedef __bf16 v16bf  __attribute__((ext_vector_type(16)));
typedef float  v8f    __attribute__((ext_vector_type(8)));

constexpr int Bc  = 2;
constexpr int Lc  = 2048;
constexpr int Dc  = 1024;
constexpr int Hc  = 16;
constexpr int DKc = 64;          // d_k
constexpr int Mc  = Bc * Lc;     // 4096 rows for the [M,K]@[K,N] GEMMs
constexpr int Nc  = Dc;
constexpr int Kc  = Dc;

__device__ inline v8f wmma_bf16(v16bf a, v16bf b, v8f c) {
  // D = A(16x32 bf16) * B(32x16 bf16) + C(16x16 f32)
  return __builtin_amdgcn_wmma_f32_16x16x32_bf16(false, a, false, b,
                                                 (short)0, c, false, false);
}

// --------------------------- async copy helpers ----------------------------
// 32-bit LDS offset of a __shared__ object (generic -> addrspace(3) -> int).
__device__ inline unsigned lds_addr(const void* p) {
  auto p3 = (__attribute__((address_space(3))) const char*)p;
  return (unsigned)(unsigned long long)p3;
}
// Per-lane 16B global -> LDS async copy; tracked by ASYNCcnt.
__device__ inline void async_copy_b128(unsigned ldsoff, const void* gsrc) {
  asm volatile("global_load_async_to_lds_b128 %0, %1, off"
               :
               : "v"(ldsoff), "v"(gsrc)
               : "memory");
}
#define S_WAIT_ASYNCCNT(N) asm volatile("s_wait_asynccnt " #N ::: "memory")

// --------------------------- WMMA fragment loads ---------------------------
// A-operand fragment (16x32 bf16), row-major [16][stride] tile in LDS.
// ISA: lane m (0..15) row m K 0-7 & 16-23; lane m+16 row m K 8-15 & 24-31.
__device__ inline v16bf load_frag_a(const __bf16* base, int stride, int lane) {
  const int lm = lane & 15, hi = lane >> 4;
  const __bf16* r = base + lm * stride;
  v16bf a;
#pragma unroll
  for (int j = 0; j < 8; ++j) a[j] = r[8 * hi + j];
#pragma unroll
  for (int j = 0; j < 8; ++j) a[8 + j] = r[16 + 8 * hi + j];
  return a;
}

// B-operand fragment (32x16 bf16), source stored TRANSPOSED in LDS as [n][k].
// ISA: lane n (0..15) col n K 0-7 & 16-23; lane n+16 col n K 8-15 & 24-31.
__device__ inline v16bf load_frag_b(const __bf16* baseT, int stride, int lane) {
  const int n = lane & 15, kh = lane >> 4;
  const __bf16* r = baseT + n * stride;
  v16bf b;
#pragma unroll
  for (int j = 0; j < 8; ++j) b[j] = r[8 * kh + j];
#pragma unroll
  for (int j = 0; j < 8; ++j) b[8 + j] = r[16 + 8 * kh + j];
  return b;
}

// ---------------------------------------------------------------------------
// fp32 -> bf16 conversion (grid-stride)
// ---------------------------------------------------------------------------
__global__ void cvt_f32_bf16(const float* __restrict__ in,
                             __bf16* __restrict__ out, int n) {
  int i = blockIdx.x * blockDim.x + threadIdx.x;
  const int stride = gridDim.x * blockDim.x;
  for (; i < n; i += stride) out[i] = (__bf16)in[i];
}

// fp32 [K,N] -> bf16 [N,K] transposed convert (LDS-tiled, 32x32 tiles).
__global__ __launch_bounds__(256) void transpose_cvt(
    const float* __restrict__ in, __bf16* __restrict__ out) {
  __shared__ float t[32][33];
  const int bx = blockIdx.x * 32, by = blockIdx.y * 32;
  const int tx = threadIdx.x, ty = threadIdx.y;  // (32, 8)
#pragma unroll
  for (int j = 0; j < 32; j += 8)
    t[ty + j][tx] = in[(size_t)(by + ty + j) * Dc + bx + tx];
  __syncthreads();
#pragma unroll
  for (int j = 0; j < 32; j += 8)
    out[(size_t)(bx + ty + j) * Dc + by + tx] = (__bf16)t[tx][ty + j];
}

// ---------------------------------------------------------------------------
// GEMM: C[M,N] = A[M,K]_bf16 @ W[K,N] + bias, with W given TRANSPOSED [N,K].
// 128 threads (4 waves), 64x64 block tile, 32x32 wave tile, K-step 32.
// Double-buffered async tile staging.
// MODE 1: store bf16 as [B,H,L,DK]   (Q,K projections)
// MODE 2: store bf16 as [B,H,DK,L]   (V projection, pre-transposed)
// MODE 3: store fp32 as [M,N]        (output projection)
// ---------------------------------------------------------------------------
constexpr int AST = 72;  // LDS row stride (bf16): 144B rows, 16B aligned
constexpr int BST = 72;

template <int MODE>
__global__ __launch_bounds__(128) void gemm64(
    const __bf16* __restrict__ A, const __bf16* __restrict__ WT,
    const float* __restrict__ bias, void* __restrict__ outp) {
  __shared__ __bf16 As[2 * 64 * AST];
  __shared__ __bf16 Bs[2 * 64 * BST];  // [n][k]

  const int tid  = threadIdx.x;
  const int lane = tid & 31, wid = tid >> 5;
  const int hi = lane >> 4, nl = lane & 15;
  const int n0 = blockIdx.x * 64, m0 = blockIdx.y * 64;
  const int wm = (wid >> 1) * 32, wn = (wid & 1) * 32;

  v8f acc[2][2];
#pragma unroll
  for (int mt = 0; mt < 2; ++mt)
#pragma unroll
    for (int nt = 0; nt < 2; ++nt)
#pragma unroll
      for (int v = 0; v < 8; ++v) acc[mt][nt][v] = 0.f;

  // 4 async b128 issues per thread per 64x32+32x64 tile pair.
  auto issue_tile = [&](int k0, int buf) {
    __bf16* as = As + buf * (64 * AST);
    __bf16* bs = Bs + buf * (64 * BST);
#pragma unroll
    for (int i = 0; i < 2; ++i) {
      const int c = tid + i * 128;
      const int r = c >> 2, c8 = (c & 3) * 8;  // 4 chunks of 8 bf16 per row
      async_copy_b128(lds_addr(&as[r * AST + c8]),
                      &A[(size_t)(m0 + r) * Kc + k0 + c8]);
      async_copy_b128(lds_addr(&bs[r * BST + c8]),
                      &WT[(size_t)(n0 + r) * Kc + k0 + c8]);
    }
  };

  issue_tile(0, 0);
  int ib = 0;
  for (int k0 = 0; k0 < Kc; k0 += 32, ib ^= 1) {
    if (k0 + 32 < Kc) {
      issue_tile(k0 + 32, ib ^ 1);
      S_WAIT_ASYNCCNT(4);  // current tile done; next tile's 4 still in flight
    } else {
      S_WAIT_ASYNCCNT(0);
    }
    __syncthreads();

    const __bf16* as = As + ib * (64 * AST);
    const __bf16* bs = Bs + ib * (64 * BST);
    v16bf a0 = load_frag_a(as + (wm + 0) * AST, AST, lane);
    v16bf a1 = load_frag_a(as + (wm + 16) * AST, AST, lane);
    v16bf b0 = load_frag_b(bs + (wn + 0) * BST, BST, lane);
    v16bf b1 = load_frag_b(bs + (wn + 16) * BST, BST, lane);
    acc[0][0] = wmma_bf16(a0, b0, acc[0][0]);
    acc[0][1] = wmma_bf16(a0, b1, acc[0][1]);
    acc[1][0] = wmma_bf16(a1, b0, acc[1][0]);
    acc[1][1] = wmma_bf16(a1, b1, acc[1][1]);
    __syncthreads();  // all waves done with buffer ib before it is refilled
  }

  // Epilogue: C layout -> global. Row = v + 8*hi, col = nl within each tile.
#pragma unroll
  for (int mt = 0; mt < 2; ++mt)
#pragma unroll
    for (int nt = 0; nt < 2; ++nt) {
      const int gn = n0 + wn + nt * 16 + nl;
      const float bb = bias[gn];
#pragma unroll
      for (int v = 0; v < 8; ++v) {
        const int gm = m0 + wm + mt * 16 + v + 8 * hi;
        const float val = acc[mt][nt][v] + bb;
        if (MODE == 3) {
          ((float*)outp)[(size_t)gm * Nc + gn] = val;
        } else {
          const int bi = gm / Lc, l = gm % Lc;
          const int hh = gn / DKc, d = gn % DKc;
          size_t idx;
          if (MODE == 1)
            idx = (((size_t)bi * Hc + hh) * Lc + l) * DKc + d;  // [B,H,L,DK]
          else
            idx = (((size_t)bi * Hc + hh) * DKc + d) * Lc + l;  // [B,H,DK,L]
          ((__bf16*)outp)[idx] = (__bf16)val;
        }
      }
    }
}

// ---------------------------------------------------------------------------
// Flash attention: one block per (q-tile of 64, b*h). 4 waves x 16 queries.
// S = Q K^T / 8, mask, online softmax, O += P V; scores never hit memory.
// K/V tiles double-buffered via async global->LDS copies.
// ---------------------------------------------------------------------------
constexpr int QST = 72, KST = 72, VST = 72, PST = 72;

__global__ __launch_bounds__(128) void flash_attn(
    const __bf16* __restrict__ Qp,   // [B,H,L,DK]
    const __bf16* __restrict__ Kp,   // [B,H,L,DK]
    const __bf16* __restrict__ Vt,   // [B,H,DK,L]  (pre-transposed)
    const int* __restrict__ mask,    // [L,L]
    __bf16* __restrict__ X) {        // [B,L,D]
  __shared__ __bf16 Qs[64 * QST];
  __shared__ __bf16 Ks[2 * 64 * KST];  // [kv][d]
  __shared__ __bf16 Vs[2 * 64 * VST];  // [d][kv]
  __shared__ __bf16 Ps[64 * PST];      // per-wave 16-row P staging

  const int tid = threadIdx.x, lane = tid & 31, wid = tid >> 5;
  const int hi = lane >> 4, nl = lane & 15;
  const int q0 = blockIdx.x * 64;
  const int bh = blockIdx.y;
  const int bb = bh >> 4, hh = bh & 15;

  const __bf16* Qg = Qp + ((size_t)bh * Lc + q0) * DKc;
  const __bf16* Kg = Kp + (size_t)bh * Lc * DKc;
  const __bf16* Vg = Vt + (size_t)bh * DKc * Lc;

  // 8 async b128 issues per thread per 64x64 K + 64x64 V tile pair.
  auto issue_kv = [&](int kv0, int buf) {
    __bf16* ks = Ks + buf * (64 * KST);
    __bf16* vs = Vs + buf * (64 * VST);
#pragma unroll
    for (int i = 0; i < 4; ++i) {
      const int c = tid + i * 128;
      const int r = c >> 3, c8 = (c & 7) * 8;  // 8 chunks of 8 bf16 per row
      async_copy_b128(lds_addr(&ks[r * KST + c8]),
                      &Kg[(size_t)(kv0 + r) * DKc + c8]);
      async_copy_b128(lds_addr(&vs[r * VST + c8]),
                      &Vg[(size_t)r * Lc + kv0 + c8]);
    }
  };

  // Prologue: Q tile (4 issues) then first K/V tile (8 issues).
#pragma unroll
  for (int i = 0; i < 4; ++i) {
    const int c = tid + i * 128;
    const int r = c >> 3, c8 = (c & 7) * 8;
    async_copy_b128(lds_addr(&Qs[r * QST + c8]), &Qg[(size_t)r * DKc + c8]);
  }
  issue_kv(0, 0);
  S_WAIT_ASYNCCNT(8);  // in-order completion => Q tile is resident
  __syncthreads();
  v16bf qa0 = load_frag_a(Qs + wid * 16 * QST + 0, QST, lane);
  v16bf qa1 = load_frag_a(Qs + wid * 16 * QST + 32, QST, lane);

  v8f o[4];
  float ms[8], ls[8];
#pragma unroll
  for (int nt = 0; nt < 4; ++nt)
#pragma unroll
    for (int v = 0; v < 8; ++v) o[nt][v] = 0.f;
#pragma unroll
  for (int v = 0; v < 8; ++v) { ms[v] = -1e30f; ls[v] = 0.f; }

  int ib = 0;
  for (int kv0 = 0; kv0 < Lc; kv0 += 64, ib ^= 1) {
    if (kv0 + 64 < Lc) {
      issue_kv(kv0 + 64, ib ^ 1);
      S_WAIT_ASYNCCNT(8);  // current K/V tile done; next still in flight
    } else {
      S_WAIT_ASYNCCNT(0);
    }
    __syncthreads();
    const __bf16* ks = Ks + ib * (64 * KST);
    const __bf16* vs = Vs + ib * (64 * VST);

    // ---- S = Q * K^T : B operand from ks[kv][d] (rows = kv = n, cols = d = k)
    v8f s[4];
#pragma unroll
    for (int nt = 0; nt < 4; ++nt) {
#pragma unroll
      for (int v = 0; v < 8; ++v) s[nt][v] = 0.f;
      v16bf bk0 = load_frag_b(ks + (nt * 16) * KST + 0, KST, lane);
      v16bf bk1 = load_frag_b(ks + (nt * 16) * KST + 32, KST, lane);
      s[nt] = wmma_bf16(qa0, bk0, s[nt]);
      s[nt] = wmma_bf16(qa1, bk1, s[nt]);
    }

    // ---- scale + mask
    const float scale = 0.125f;  // 1/sqrt(64)
#pragma unroll
    for (int nt = 0; nt < 4; ++nt)
#pragma unroll
      for (int v = 0; v < 8; ++v) {
        float sv = s[nt][v] * scale;
        const int gq = q0 + wid * 16 + v + 8 * hi;
        const int gk = kv0 + nt * 16 + nl;
        if (mask[(size_t)gq * Lc + gk] == 0) sv = -1e9f;
        s[nt][v] = sv;
      }

    // ---- online softmax (row stats per C-layout row; reduce over 16 lanes)
#pragma unroll
    for (int v = 0; v < 8; ++v) {
      float mx = fmaxf(fmaxf(s[0][v], s[1][v]), fmaxf(s[2][v], s[3][v]));
      mx = fmaxf(mx, __shfl_xor(mx, 1));
      mx = fmaxf(mx, __shfl_xor(mx, 2));
      mx = fmaxf(mx, __shfl_xor(mx, 4));
      mx = fmaxf(mx, __shfl_xor(mx, 8));
      const float mnew = fmaxf(ms[v], mx);
      const float alpha = __expf(ms[v] - mnew);
      float rs = 0.f;
#pragma unroll
      for (int nt = 0; nt < 4; ++nt) {
        const float p = __expf(s[nt][v] - mnew);
        s[nt][v] = p;
        rs += p;
      }
      rs += __shfl_xor(rs, 1);
      rs += __shfl_xor(rs, 2);
      rs += __shfl_xor(rs, 4);
      rs += __shfl_xor(rs, 8);
      ls[v] = ls[v] * alpha + rs;
      ms[v] = mnew;
#pragma unroll
      for (int nt = 0; nt < 4; ++nt) o[nt][v] *= alpha;
    }

    // ---- P (C layout) -> per-wave LDS -> A-operand layout
    __bf16* ps = Ps + wid * 16 * PST;
#pragma unroll
    for (int nt = 0; nt < 4; ++nt)
#pragma unroll
      for (int v = 0; v < 8; ++v)
        ps[(v + 8 * hi) * PST + nt * 16 + nl] = (__bf16)s[nt][v];

    v16bf pa0 = load_frag_a(ps + 0, PST, lane);   // kv 0..31
    v16bf pa1 = load_frag_a(ps + 32, PST, lane);  // kv 32..63

    // ---- O += P * V : B operand from vs[d][kv] (rows = d = n, cols = kv = k)
#pragma unroll
    for (int nt = 0; nt < 4; ++nt) {
      v16bf bv0 = load_frag_b(vs + (nt * 16) * VST + 0, VST, lane);
      v16bf bv1 = load_frag_b(vs + (nt * 16) * VST + 32, VST, lane);
      o[nt] = wmma_bf16(pa0, bv0, o[nt]);
      o[nt] = wmma_bf16(pa1, bv1, o[nt]);
    }
    __syncthreads();  // all waves done with buffer ib before it is refilled
  }

  // ---- normalize and write x = [B,L,D] (head-concat layout, bf16)
#pragma unroll
  for (int v = 0; v < 8; ++v) ls[v] = 1.0f / ls[v];
#pragma unroll
  for (int nt = 0; nt < 4; ++nt)
#pragma unroll
    for (int v = 0; v < 8; ++v) {
      const int gq = q0 + wid * 16 + v + 8 * hi;
      const int d = nt * 16 + nl;
      X[((size_t)bb * Lc + gq) * Dc + hh * DKc + d] = (__bf16)(o[nt][v] * ls[v]);
    }
}

// ---------------------------------------------------------------------------
extern "C" void kernel_launch(void* const* d_in, const int* in_sizes, int n_in,
                              void* d_out, int out_size, void* d_ws,
                              size_t ws_size, hipStream_t stream) {
  (void)in_sizes; (void)n_in; (void)out_size; (void)ws_size;
  const float* q  = (const float*)d_in[0];
  const float* k  = (const float*)d_in[1];
  const float* v  = (const float*)d_in[2];
  const float* Wq = (const float*)d_in[3];
  const float* bq = (const float*)d_in[4];
  const float* Wk = (const float*)d_in[5];
  const float* bk = (const float*)d_in[6];
  const float* Wv = (const float*)d_in[7];
  const float* bv = (const float*)d_in[8];
  const float* Wo = (const float*)d_in[9];
  const float* bo = (const float*)d_in[10];
  const int* mask = (const int*)d_in[11];
  float* out = (float*)d_out;

  const size_t BIG = (size_t)Mc * Dc;  // 4Mi elements
  const size_t WSZ = (size_t)Dc * Dc;  // 1Mi elements
  __bf16* qb  = (__bf16*)d_ws;   // bf16 query   (later reused as X)
  __bf16* kb  = qb + BIG;        // bf16 key     (later reused as Vt)
  __bf16* vb  = kb + BIG;        // bf16 value
  __bf16* Qp  = vb + BIG;        // projected Q [B,H,L,DK]
  __bf16* Kp  = Qp + BIG;        // projected K [B,H,L,DK]
  __bf16* wqb = Kp + BIG;        // W^T bf16 [N,K]
  __bf16* wkb = wqb + WSZ;
  __bf16* wvb = wkb + WSZ;
  __bf16* wob = wvb + WSZ;
  __bf16* Vt = kb;  // alias: kb free after GEMM-K
  __bf16* X  = qb;  // alias: qb free after GEMM-Q

  // 1) fp32 -> bf16 (inputs) and fp32 -> bf16 transposed (weights)
  cvt_f32_bf16<<<1024, 256, 0, stream>>>(q, qb, (int)BIG);
  cvt_f32_bf16<<<1024, 256, 0, stream>>>(k, kb, (int)BIG);
  cvt_f32_bf16<<<1024, 256, 0, stream>>>(v, vb, (int)BIG);
  dim3 tgrid(Dc / 32, Dc / 32), tblk(32, 8);
  transpose_cvt<<<tgrid, tblk, 0, stream>>>(Wq, wqb);
  transpose_cvt<<<tgrid, tblk, 0, stream>>>(Wk, wkb);
  transpose_cvt<<<tgrid, tblk, 0, stream>>>(Wv, wvb);
  transpose_cvt<<<tgrid, tblk, 0, stream>>>(Wo, wob);

  // 2) projections
  dim3 ggrid(Nc / 64, Mc / 64);  // (16, 64)
  gemm64<1><<<ggrid, 128, 0, stream>>>(qb, wqb, bq, (void*)Qp);
  gemm64<1><<<ggrid, 128, 0, stream>>>(kb, wkb, bk, (void*)Kp);
  gemm64<2><<<ggrid, 128, 0, stream>>>(vb, wvb, bv, (void*)Vt);

  // 3) flash attention
  dim3 agrid(Lc / 64, Bc * Hc);  // (32, 32)
  flash_attn<<<agrid, 128, 0, stream>>>(Qp, Kp, Vt, mask, X);

  // 4) output projection (fp32 result)
  gemm64<3><<<ggrid, 128, 0, stream>>>(X, wob, bo, (void*)out);
}
